// DeepLearningRecommenderModel_89824946029305
// MI455X (gfx1250) — matrix operand order repaired and checked by reading, not compiled, not run on hardware
//
#include <hip/hip_runtime.h>
#include <hip/hip_bf16.h>

typedef __attribute__((ext_vector_type(16))) _Float16 v16h;
typedef __attribute__((ext_vector_type(8)))  _Float16 v8h;
typedef __attribute__((ext_vector_type(8)))  float    v8f;

#define BATCH   16384
#define NFEAT   26
#define DEMB    64
#define VOCAB   100000
#define NDENSE  13
#define NEMB    27            // 1 dense + 26 sparse rows
#define ZDIM    2079          // 27*64 + 351
#define ZPAD    2080          // pad K to multiple of 32

// ---------------------------------------------------------------------------
// Fragment helpers (CDNA5 WMMA 16x16x32 f16 layouts, ISA 7.12.2)
// A 16x32 f16: lane half=lane>>4, row=lane&15.
//   v16h elems 0..7  = K (half*16 + 0..7)        [VGPR0..3]
//   v16h elems 8..15 = K (half*16 + 8..15)   [VGPR4..7]
// B 32x16 f16 (loaded from Bt = B^T row-major): lane col = lane&15,
//   elems e -> K = (lane>>4)*16 + e  (16 contiguous f16)
// C/D 16x16 f32: VGPR r, lane L: m = r + (L>>4)*8, n = L&15
// ---------------------------------------------------------------------------
__device__ inline v16h load_a_frag(const _Float16* __restrict__ arow, int k0, int half) {
    v8h lo = *(const v8h*)(arow + k0 + half * 16);
    v8h hi = *(const v8h*)(arow + k0 + half * 16 + 8);
    v16h a;
#pragma unroll
    for (int e = 0; e < 8; ++e) { a[e] = lo[e]; a[e + 8] = hi[e]; }
    return a;
}

__device__ inline v8h cvt8(float4 f0, float4 f1) {
    v8h h;
    h[0] = (_Float16)f0.x; h[1] = (_Float16)f0.y;
    h[2] = (_Float16)f0.z; h[3] = (_Float16)f0.w;
    h[4] = (_Float16)f1.x; h[5] = (_Float16)f1.y;
    h[6] = (_Float16)f1.z; h[7] = (_Float16)f1.w;
    return h;
}

// Async global->LDS copy of 16 bytes per lane (CDNA5 async-DMA path).
__device__ inline void async_copy_b128(const void* gsrc, void* lds_dst) {
    uint64_t g = (uint64_t)(uintptr_t)gsrc;
    uint32_t l = (uint32_t)(uintptr_t)lds_dst;
    asm volatile("global_load_async_to_lds_b128 %0, %1, off"
                 :: "v"(l), "v"(g) : "memory");
}

// ---------------------------------------------------------------------------
// Weight convert+transpose: wt[n][kp] = (kp < K) ? w[kp][n] : 0   (f32 -> f16)
// ---------------------------------------------------------------------------
__global__ void __launch_bounds__(256) convert_w_t(const float* __restrict__ w,
                                                   _Float16* __restrict__ wt,
                                                   int K, int N, int Kpad) {
    long idx = (long)blockIdx.x * blockDim.x + threadIdx.x;
    long total = (long)N * Kpad;
    if (idx >= total) return;
    int n = (int)(idx / Kpad);
    int kp = (int)(idx % Kpad);
    wt[idx] = (kp < K) ? (_Float16)w[(long)kp * N + n] : (_Float16)0.0f;
}

// ---------------------------------------------------------------------------
// Bottom MLP: dense (B x 13) -> relu(x@dw1+db1) -> relu(h@dw2+db2) (B x 64)
// 64 threads per sample, 4 samples per 256-thread block.
// ---------------------------------------------------------------------------
__global__ void __launch_bounds__(256) bottom_mlp(const float* __restrict__ dense,
                                                  const float* __restrict__ dw1,
                                                  const float* __restrict__ db1,
                                                  const float* __restrict__ dw2,
                                                  const float* __restrict__ db2,
                                                  float* __restrict__ dense_emb) {
    __shared__ float hsh[4][DEMB];
    const int j   = threadIdx.x & 63;
    const int rib = threadIdx.x >> 6;
    const int b   = blockIdx.x * 4 + rib;
    float acc = db1[j];
#pragma unroll
    for (int k = 0; k < NDENSE; ++k)
        acc += dense[(size_t)b * NDENSE + k] * dw1[k * DEMB + j];
    hsh[rib][j] = fmaxf(acc, 0.0f);
    __syncthreads();
    float acc2 = db2[j];
#pragma unroll 8
    for (int k = 0; k < DEMB; ++k)
        acc2 += hsh[rib][k] * dw2[k * DEMB + j];
    dense_emb[(size_t)b * DEMB + j] = fmaxf(acc2, 0.0f);
}

// ---------------------------------------------------------------------------
// Gather + interaction + z assembly. One wave per sample, 8 samples/block.
// LDS: 32x64 f16 emb tile per wave. inter = emb @ emb^T via WMMA; write
// flattened emb (27*64) then 351 upper-triangle pairs into z (f16, ZPAD).
// Gather is the HBM-bound stage (~109 MB) -> full-width b128 loads.
// ---------------------------------------------------------------------------
__global__ void __launch_bounds__(256) interact_kernel(const float* __restrict__ dense_emb,
                                                       const int* __restrict__ sparse_idx,
                                                       const float* __restrict__ emb_tables,
                                                       _Float16* __restrict__ z) {
    __shared__ _Float16 esh[8][32][DEMB];
    const int lane = threadIdx.x & 31;
    const int wave = threadIdx.x >> 5;
    const int b    = blockIdx.x * 8 + wave;

    // Stage: lane r owns row r of the 32x64 tile (rows 27..31 zero pad)
    {
        const int row = lane;
        if (row < NEMB) {
            const float* s = (row == 0)
                ? dense_emb + (size_t)b * DEMB
                : emb_tables + ((size_t)(row - 1) * VOCAB +
                                sparse_idx[(size_t)b * NFEAT + (row - 1)]) * DEMB;
            const float4* s4 = (const float4*)s;
            _Float16* zrow = z + (size_t)b * ZPAD + row * DEMB;
#pragma unroll
            for (int c = 0; c < 8; ++c) {
                v8h h = cvt8(s4[2 * c], s4[2 * c + 1]);
                *(v8h*)&esh[wave][row][c * 8] = h;
                *(v8h*)&zrow[c * 8] = h;
            }
        } else {
            v8h zeroh;
#pragma unroll
            for (int e = 0; e < 8; ++e) zeroh[e] = (_Float16)0.0f;
#pragma unroll
            for (int c = 0; c < 8; ++c)
                *(v8h*)&esh[wave][row][c * 8] = zeroh;
            if (row == 31) z[(size_t)b * ZPAD + (ZPAD - 1)] = (_Float16)0.0f;
        }
    }
    __syncthreads();

    const int half = lane >> 4;
    const int l15  = lane & 15;

    // Upper-triangle tiles of the 32x32 Gram matrix: (0,0), (0,1), (1,1)
#pragma unroll
    for (int t = 0; t < 3; ++t) {
        const int m0 = (t == 2) ? 16 : 0;
        const int n0 = (t == 0) ? 0 : 16;
        v8f acc;
#pragma unroll
        for (int r = 0; r < 8; ++r) acc[r] = 0.0f;
#pragma unroll
        for (int k0 = 0; k0 < DEMB; k0 += 32) {
            v16h a  = load_a_frag(&esh[wave][m0 + l15][0], k0, half);
            v16h bf = *(const v16h*)(&esh[wave][n0 + l15][0] + k0 + half * 16);
            acc = __builtin_amdgcn_wmma_f32_16x16x32_f16(
                false, a, false, bf, (short)0, acc, false, false);
        }
#pragma unroll
        for (int r = 0; r < 8; ++r) {
            const int m = m0 + half * 8 + r;
            const int n = n0 + l15;
            if (m < n && n < NEMB) {
                // triu(27, k=1) row-major pair index
                const int p = m * (2 * NEMB - m - 1) / 2 + (n - m - 1);
                z[(size_t)b * ZPAD + NEMB * DEMB + p] = (_Float16)acc[r];
            }
        }
    }
}

// ---------------------------------------------------------------------------
// WMMA GEMM with async-DMA B staging: C = relu(A @ Bt^T + bias).
// Block tile = 256(M) x 64(N): 8 waves, each a 32x64 sub-tile with 8
// accumulators (2 M-tiles x 4 N-tiles). The 64x32 f16 B panel per k-chunk is
// shared by all 8 waves -> staged once per block into LDS with
// global_load_async_to_lds_b128, double-buffered, s_wait_asynccnt + barrier.
// Requires: M % 256 == 0, N % 64 == 0, K % 32 == 0 (all true here).
// ---------------------------------------------------------------------------
__global__ void __launch_bounds__(256) wmma_gemm_bias_relu(const _Float16* __restrict__ A,
                                                           const _Float16* __restrict__ Bt,
                                                           const float* __restrict__ bias,
                                                           _Float16* __restrict__ Cout,
                                                           int M, int N, int K,
                                                           int lda, int ldb, int ldc) {
    __shared__ _Float16 bsh[2][64][32];      // double-buffered B panel (8 KB)
    const int tid  = threadIdx.x;
    const int lane = tid & 31;
    const int wave = tid >> 5;
    const int nstrips = N >> 6;
    const int n0 = (blockIdx.x % nstrips) << 6;
    const int m0 = ((blockIdx.x / nstrips) << 8) + (wave << 5);
    const int half = lane >> 4;
    const int l15  = lane & 15;

    // Async staging assignment: thread t copies 16 B of row (t>>2), seg (t&3).
    const int srow = tid >> 2;
    const int sseg = tid & 3;
    const _Float16* bsrc = Bt + (size_t)(n0 + srow) * ldb + sseg * 8;

    v8f acc[2][4];
#pragma unroll
    for (int j = 0; j < 4; ++j) {
        const float bv = bias[n0 + 16 * j + l15];
#pragma unroll
        for (int i = 0; i < 2; ++i)
#pragma unroll
            for (int r = 0; r < 8; ++r) acc[i][j][r] = bv;
    }

    const _Float16* arow0 = A + (size_t)(m0 + l15) * lda;
    const _Float16* arow1 = A + (size_t)(m0 + 16 + l15) * lda;

    const int KC = K >> 5;
    async_copy_b128(bsrc, &bsh[0][srow][sseg * 8]);          // prefill chunk 0

    for (int kc = 0; kc < KC; ++kc) {
        const int kb = kc << 5;
        __syncthreads();                     // everyone done reading buf^1
        if (kc + 1 < KC) {
            async_copy_b128(bsrc + (kb + 32), &bsh[(kc + 1) & 1][srow][sseg * 8]);
            asm volatile("s_wait_asynccnt 0x1" ::: "memory");   // chunk kc landed
        } else {
            asm volatile("s_wait_asynccnt 0x0" ::: "memory");
        }
        __syncthreads();                     // panel visible to all waves
        const int buf = kc & 1;

        __builtin_prefetch(arow0 + kb + 64, 0, 0);   // global_prefetch_b8
        __builtin_prefetch(arow1 + kb + 64, 0, 0);
        v16h a0 = load_a_frag(arow0, kb, half);
        v16h a1 = load_a_frag(arow1, kb, half);
#pragma unroll
        for (int j = 0; j < 4; ++j) {
            v16h bf = *(const v16h*)(&bsh[buf][16 * j + l15][half * 16]);
            acc[0][j] = __builtin_amdgcn_wmma_f32_16x16x32_f16(
                false, a0, false, bf, (short)0, acc[0][j], false, false);
            acc[1][j] = __builtin_amdgcn_wmma_f32_16x16x32_f16(
                false, a1, false, bf, (short)0, acc[1][j], false, false);
        }
    }

#pragma unroll
    for (int i = 0; i < 2; ++i)
#pragma unroll
        for (int j = 0; j < 4; ++j)
#pragma unroll
            for (int r = 0; r < 8; ++r) {
                const int m = m0 + 16 * i + half * 8 + r;
                const float v = fmaxf(acc[i][j][r], 0.0f);
                Cout[(size_t)m * ldc + n0 + 16 * j + l15] = (_Float16)v;
            }
}

// ---------------------------------------------------------------------------
// Final 128 -> 1 layer, f32 accumulation.
// ---------------------------------------------------------------------------
__global__ void __launch_bounds__(256) final_layer(const _Float16* __restrict__ a3,
                                                   const float* __restrict__ w4,
                                                   const float* __restrict__ b4,
                                                   float* __restrict__ out) {
    const int b = blockIdx.x * blockDim.x + threadIdx.x;
    float acc = b4[0];
#pragma unroll 8
    for (int k = 0; k < 128; ++k)
        acc += (float)a3[(size_t)b * 128 + k] * w4[k];
    out[b] = acc;
}

// ---------------------------------------------------------------------------
extern "C" void kernel_launch(void* const* d_in, const int* in_sizes, int n_in,
                              void* d_out, int out_size, void* d_ws, size_t ws_size,
                              hipStream_t stream) {
    const float* dense      = (const float*)d_in[0];
    const int*   sparse_idx = (const int*)d_in[1];
    const float* emb_tables = (const float*)d_in[2];
    const float* dw1 = (const float*)d_in[3];
    const float* db1 = (const float*)d_in[4];
    const float* dw2 = (const float*)d_in[5];
    const float* db2 = (const float*)d_in[6];
    const float* ow1 = (const float*)d_in[7];
    const float* ob1 = (const float*)d_in[8];
    const float* ow2 = (const float*)d_in[9];
    const float* ob2 = (const float*)d_in[10];
    const float* ow3 = (const float*)d_in[11];
    const float* ob3 = (const float*)d_in[12];
    const float* ow4 = (const float*)d_in[13];
    const float* ob4 = (const float*)d_in[14];
    float* out = (float*)d_out;

    char* ws = (char*)d_ws;
    size_t off = 0;
    auto alloc = [&](size_t bytes) -> void* {
        void* p = ws + off;
        off = (off + bytes + 255) & ~(size_t)255;
        return p;
    };

    _Float16* w1t  = (_Float16*)alloc((size_t)512 * ZPAD * sizeof(_Float16));
    _Float16* w2t  = (_Float16*)alloc((size_t)256 * 512 * sizeof(_Float16));
    _Float16* w3t  = (_Float16*)alloc((size_t)128 * 256 * sizeof(_Float16));
    float*    demb = (float*)   alloc((size_t)BATCH * DEMB * sizeof(float));
    _Float16* z    = (_Float16*)alloc((size_t)BATCH * ZPAD * sizeof(_Float16));
    _Float16* a1   = (_Float16*)alloc((size_t)BATCH * 512 * sizeof(_Float16));
    _Float16* a2   = (_Float16*)alloc((size_t)BATCH * 256 * sizeof(_Float16));
    _Float16* a3   = (_Float16*)alloc((size_t)BATCH * 128 * sizeof(_Float16));
    (void)ws_size; (void)in_sizes; (void)n_in; (void)out_size;

    // Weight transposes / f16 conversion (tiny; weights reused across batch)
    {
        long t1 = 512L * ZPAD;
        convert_w_t<<<(int)((t1 + 255) / 256), 256, 0, stream>>>(ow1, w1t, ZDIM, 512, ZPAD);
        long t2 = 256L * 512;
        convert_w_t<<<(int)((t2 + 255) / 256), 256, 0, stream>>>(ow2, w2t, 512, 256, 512);
        long t3 = 128L * 256;
        convert_w_t<<<(int)((t3 + 255) / 256), 256, 0, stream>>>(ow3, w3t, 256, 128, 256);
    }

    // Bottom MLP (f32 VALU, 161 MFLOP)
    bottom_mlp<<<BATCH / 4, 256, 0, stream>>>(dense, dw1, db1, dw2, db2, demb);

    // Gather + interaction (WMMA) + z assembly — HBM-bound (~109 MB gather)
    interact_kernel<<<BATCH / 8, 256, 0, stream>>>(demb, sparse_idx, emb_tables, z);

    // Top MLP: three async-staged, register-blocked WMMA GEMMs (bias+relu)
    {
        int blocks = (BATCH / 256) * (512 / 64);
        wmma_gemm_bias_relu<<<blocks, 256, 0, stream>>>(
            z, w1t, ob1, a1, BATCH, 512, ZPAD, ZPAD, ZPAD, 512);
    }
    {
        int blocks = (BATCH / 256) * (256 / 64);
        wmma_gemm_bias_relu<<<blocks, 256, 0, stream>>>(
            a1, w2t, ob2, a2, BATCH, 256, 512, 512, 512, 256);
    }
    {
        int blocks = (BATCH / 256) * (128 / 64);
        wmma_gemm_bias_relu<<<blocks, 256, 0, stream>>>(
            a2, w3t, ob3, a3, BATCH, 128, 256, 256, 256, 128);
    }

    // Final 128 -> 1
    final_layer<<<BATCH / 256, 256, 0, stream>>>(a3, ow4, ob4, out);
}
